// MorphologyNet_5145370820913
// MI455X (gfx1250) — matrix operand order repaired and checked by reading, not compiled, not run on hardware
//
#include <hip/hip_runtime.h>
#include <hip/hip_bf16.h>
#include <cmath>

typedef __attribute__((ext_vector_type(16))) _Float16 v16h;
typedef __attribute__((ext_vector_type(8)))  float    v8f;

#define HW      65536      // 256*256
#define NTOT    4194304    // 4*16*65536

struct G5 { float g[5]; };

__device__ __forceinline__ int refl(int i) {
    i = (i < 0) ? -i : i;
    return (i > 255) ? (510 - i) : i;
}
__device__ __forceinline__ float sigm(float x) { return 1.0f / (1.0f + expf(-x)); }

// CDNA5 async global->LDS copy (ASYNCcnt-tracked), 16B per lane per issue.
__device__ __forceinline__ void async_copy_b128(unsigned int lds_off,
                                                const void* gptr) {
    unsigned long long ga = (unsigned long long)gptr;
    asm volatile("global_load_async_to_lds_b128 %0, %1, off"
                 :: "v"(lds_off), "v"(ga) : "memory");
}
__device__ __forceinline__ void wait_async_all() {
    asm volatile("s_wait_asynccnt 0x0" ::: "memory");
}

// ---------------------------------------------------------------- weight prep
// W1 [16,112] fp32 -> W1h [16,128] f16 (zero-padded K); W2 [16,32] -> W2h f16
__global__ void __launch_bounds__(256)
prep_weights_kernel(const float* __restrict__ W1, const float* __restrict__ W2,
                    _Float16* __restrict__ W1h, _Float16* __restrict__ W2h) {
    int i = blockIdx.x * 256 + threadIdx.x;
    if (i < 2048) {
        int o = i >> 7, k = i & 127;
        W1h[i] = (_Float16)((k < 112) ? W1[o * 112 + k] : 0.0f);
    } else if (i < 2560) {
        int j = i - 2048;
        W2h[j] = (_Float16)W2[j];
    }
}

// x [b][c][hw] fp32 -> combH channels 0..15 f16; zero channels 112..127
__global__ void __launch_bounds__(256)
prep_x_kernel(const float* __restrict__ x, _Float16* __restrict__ combH) {
    int idx = blockIdx.x * 256 + threadIdx.x;
    int plane = idx >> 16, hw = idx & 65535;
    int b = plane >> 4, c = plane & 15;
    combH[((size_t)b * 128 + c) * HW + hw]       = (_Float16)x[idx];
    combH[((size_t)b * 128 + 112 + c) * HW + hw] = (_Float16)0.0f;
}

// ---------------------------------------------------------------- morphology
__global__ void __launch_bounds__(256)
erode_kernel(const float* __restrict__ x, float* __restrict__ out, int k) {
    int idx = blockIdx.x * 256 + threadIdx.x;
    int plane = idx >> 16, hw = idx & 65535, hh = hw >> 8, ww = hw & 255;
    int p = k >> 1;
    const float* s = x + (size_t)plane * HW;
    float mn = 3.4028235e38f;
    for (int dy = -p; dy <= p; ++dy) {
        int y = refl(hh + dy) * 256;
        for (int dx = -p; dx <= p; ++dx)
            mn = fminf(mn, s[y + refl(ww + dx)]);
    }
    out[idx] = mn;
}

// dilate(eroded) -> top-hat -> affine+relu -> f16 combined channel block
__global__ void __launch_bounds__(256)
tophat_kernel(const float* __restrict__ x, const float* __restrict__ eroded,
              const float* __restrict__ ts, const float* __restrict__ tb,
              _Float16* __restrict__ combH, int scaleIdx, int k) {
    int idx = blockIdx.x * 256 + threadIdx.x;
    int plane = idx >> 16, hw = idx & 65535, hh = hw >> 8, ww = hw & 255;
    int b = plane >> 4, c = plane & 15;
    int p = k >> 1;
    const float* s = eroded + (size_t)plane * HW;
    float mx = -3.4028235e38f;
    for (int dy = -p; dy <= p; ++dy) {
        int y = refl(hh + dy) * 256;
        for (int dx = -p; dx <= p; ++dx)
            mx = fmaxf(mx, s[y + refl(ww + dx)]);
    }
    float th = x[idx] - mx;
    float v  = fmaxf(ts[scaleIdx * 16 + c] * th + tb[scaleIdx * 16 + c], 0.0f);
    combH[((size_t)b * 128 + 16 + scaleIdx * 16 + c) * HW + hw] = (_Float16)v;
}

// ---------------------------------------------------------------- LoG path
__global__ void __launch_bounds__(256)
gaussh_kernel(const float* __restrict__ x, float* __restrict__ out, G5 G) {
    int idx = blockIdx.x * 256 + threadIdx.x;
    int plane = idx >> 16, hw = idx & 65535, hh = hw >> 8, ww = hw & 255;
    const float* s = x + (size_t)plane * HW + hh * 256;
    float a = 0.0f;
    #pragma unroll
    for (int d = -2; d <= 2; ++d) a += G.g[d + 2] * s[refl(ww + d)];
    out[idx] = a;
}

__global__ void __launch_bounds__(256)
gaussv_kernel(const float* __restrict__ in, float* __restrict__ out, G5 G) {
    int idx = blockIdx.x * 256 + threadIdx.x;
    int plane = idx >> 16, hw = idx & 65535, hh = hw >> 8, ww = hw & 255;
    const float* s = in + (size_t)plane * HW;
    float a = 0.0f;
    #pragma unroll
    for (int d = -2; d <= 2; ++d) a += G.g[d + 2] * s[refl(hh + d) * 256 + ww];
    out[idx] = a;
}

__global__ void __launch_bounds__(256)
log_kernel(const float* __restrict__ sm, const float* __restrict__ pw,
           const float* __restrict__ nw, _Float16* __restrict__ combH, int scaleIdx) {
    int idx = blockIdx.x * 256 + threadIdx.x;
    int plane = idx >> 16, hw = idx & 65535, hh = hw >> 8, ww = hw & 255;
    int b = plane >> 4, c = plane & 15;
    const float* s = sm + (size_t)plane * HW;
    float lo = 4.0f * s[hh * 256 + ww]
             - s[refl(hh - 1) * 256 + ww] - s[refl(hh + 1) * 256 + ww]
             - s[hh * 256 + refl(ww - 1)] - s[hh * 256 + refl(ww + 1)];
    float v = fmaxf(lo, 0.0f) * pw[scaleIdx * 16 + c]
            + fmaxf(-lo, 0.0f) * nw[scaleIdx * 16 + c];
    combH[((size_t)b * 128 + 64 + scaleIdx * 16 + c) * HW + hw] = (_Float16)v;
}

// ------------------------------------------------- W1 1x1 conv GEMM via WMMA
// enhanced[16,NP] = relu(W1h[16x128] * combH[128,NP])
// Block: 128 thr = 4 waves, 64 pixels.  B tile (128ch x 64px f16 = 16KB) is
// staged to LDS with CDNA5 async global->LDS B128 copies, then gathered into
// WMMA fragments.  LDS row pitch 72 halves (144B) keeps B128 targets 16B
// aligned.
__global__ void __launch_bounds__(128)
w1_gemm_wmma(const _Float16* __restrict__ combH, const _Float16* __restrict__ W1h,
             _Float16* __restrict__ enhH) {
    __shared__ __align__(16) _Float16 tile[128 * 72];   // 18432 B

    const int tid  = threadIdx.x;
    const int lane = tid & 31;
    const int wid  = tid >> 5;
    const int pixB = blockIdx.x * 64;        // block pixel base (64 | 65536)
    const int b    = pixB >> 16;
    const int hwB  = pixB & 65535;
    const int m    = lane & 15;              // A row == B/D column (pixel)
    const int h    = lane >> 4;

    // ---- stage B tile: 128 rows x 8 chunks of 16B, 8 async issues/thread
    const _Float16* gbase = combH + (size_t)b * 128 * HW + hwB;
    #pragma unroll
    for (int it = 0; it < 8; ++it) {
        int chunk = it * 128 + tid;          // 0..1023
        int ch = chunk >> 3, c8 = chunk & 7;
        unsigned int loff = (unsigned int)(size_t)&tile[ch * 72 + c8 * 8];
        async_copy_b128(loff, gbase + (size_t)ch * HW + c8 * 8);
    }
    wait_async_all();
    __syncthreads();

    // ---- A fragments + 4 WMMA k-steps
    v8f acc = {0.f, 0.f, 0.f, 0.f, 0.f, 0.f, 0.f, 0.f};
    const int wpx = wid * 16;                // this wave's pixel offset in tile
    #pragma unroll
    for (int ks = 0; ks < 4; ++ks) {
        v16h afr, bfr;
        #pragma unroll
        for (int j = 0; j < 16; ++j) {
            int kA = ks * 32 + ((j < 8) ? (8 * h + j) : (16 + 8 * h + (j - 8)));
            afr[j] = W1h[m * 128 + kA];
            int row = ks * 32 + ((j < 8) ? (8 * h + j) : (16 + 8 * h + (j - 8)));
            bfr[j] = tile[row * 72 + wpx + m];
        }
        acc = __builtin_amdgcn_wmma_f32_16x16x32_f16(false, afr, false, bfr,
                                                     (short)0, acc, false, false);
    }
    #pragma unroll
    for (int r = 0; r < 8; ++r) {
        int o = r + 8 * h;
        float v = fmaxf(acc[r], 0.0f);
        enhH[((size_t)b * 32 + o) * HW + hwB + wpx + m] = (_Float16)v;
    }
}

// ---------------------------------------------------- soft max-median filter
__global__ void __launch_bounds__(256)
median_kernel(const _Float16* __restrict__ ed, const float* __restrict__ mix,
              const float* __restrict__ beta_raw, _Float16* __restrict__ edOut) {
    int idx = blockIdx.x * 256 + threadIdx.x;
    int plane = idx >> 16, hw = idx & 65535, hh = hw >> 8, ww = hw & 255;
    int b = plane >> 4, c = plane & 15;
    const _Float16* s = ed + ((size_t)b * 32 + c) * HW;
    float v[25];
    float mx = -3.4028235e38f, mean = 0.0f;
    int q = 0;
    #pragma unroll
    for (int dy = -2; dy <= 2; ++dy) {
        int y = refl(hh + dy) * 256;
        #pragma unroll
        for (int dx = -2; dx <= 2; ++dx) {
            float t = (float)s[y + refl(ww + dx)];
            v[q++] = t;
            mx = fmaxf(mx, t);
            mean += t;
        }
    }
    mean *= (1.0f / 25.0f);
    float beta = 5.0f + 45.0f * sigm(beta_raw[0]);
    float sw = 0.0f, swv = 0.0f;
    #pragma unroll
    for (int i = 0; i < 25; ++i) {
        float w = expf(-beta * fabsf(v[i] - mean));
        sw += w; swv += w * v[i];
    }
    float med = swv / sw;
    float lam = sigm(mix[c]);
    float den = lam * med + (1.0f - lam) * mx;
    edOut[((size_t)b * 32 + 16 + c) * HW + hw] = (_Float16)den;
}

// ------------------------------------------------- W2 1x1 conv GEMM via WMMA
// refined[16,NP] = relu(W2h[16x32] * ed[32,NP]); async-staged 32x64 B tile.
__global__ void __launch_bounds__(128)
w2_gemm_wmma(const _Float16* __restrict__ ed, const _Float16* __restrict__ W2h,
             float* __restrict__ refined) {
    __shared__ __align__(16) _Float16 tile[32 * 72];    // 4608 B

    const int tid  = threadIdx.x;
    const int lane = tid & 31;
    const int wid  = tid >> 5;
    const int pixB = blockIdx.x * 64;
    const int b    = pixB >> 16;
    const int hwB  = pixB & 65535;
    const int m    = lane & 15;
    const int h    = lane >> 4;

    const _Float16* gbase = ed + (size_t)b * 32 * HW + hwB;
    #pragma unroll
    for (int it = 0; it < 2; ++it) {
        int chunk = it * 128 + tid;          // 0..255
        int ch = chunk >> 3, c8 = chunk & 7;
        unsigned int loff = (unsigned int)(size_t)&tile[ch * 72 + c8 * 8];
        async_copy_b128(loff, gbase + (size_t)ch * HW + c8 * 8);
    }
    wait_async_all();
    __syncthreads();

    const int wpx = wid * 16;
    v16h afr, bfr;
    #pragma unroll
    for (int j = 0; j < 16; ++j) {
        int kA = (j < 8) ? (8 * h + j) : (16 + 8 * h + (j - 8));
        afr[j] = W2h[m * 32 + kA];
        int row = (j < 8) ? (8 * h + j) : (16 + 8 * h + (j - 8));
        bfr[j] = tile[row * 72 + wpx + m];
    }
    v8f acc = {0.f, 0.f, 0.f, 0.f, 0.f, 0.f, 0.f, 0.f};
    acc = __builtin_amdgcn_wmma_f32_16x16x32_f16(false, afr, false, bfr,
                                                 (short)0, acc, false, false);
    #pragma unroll
    for (int r = 0; r < 8; ++r) {
        int o = r + 8 * h;
        refined[((size_t)b * 16 + o) * HW + hwB + wpx + m] = fmaxf(acc[r], 0.0f);
    }
}

// -------------------------------------------- per-(b,c) logsumexp reduction
__global__ void __launch_bounds__(256)
lse_kernel(const float* __restrict__ refined, float* __restrict__ maxval) {
    __shared__ float sd[256];
    int tid = threadIdx.x;
    int q = blockIdx.x;                 // plane = b*16 + c
    const float* r = refined + (size_t)q * HW;
    float m = -3.4028235e38f;
    for (int i = tid; i < HW; i += 256) m = fmaxf(m, r[i]);
    sd[tid] = m;
    __syncthreads();
    for (int s = 128; s > 0; s >>= 1) {
        if (tid < s) sd[tid] = fmaxf(sd[tid], sd[tid + s]);
        __syncthreads();
    }
    float m10 = 10.0f * sd[0];
    __syncthreads();
    float sum = 0.0f;
    for (int i = tid; i < HW; i += 256) sum += expf(10.0f * r[i] - m10);
    sd[tid] = sum;
    __syncthreads();
    for (int s = 128; s > 0; s >>= 1) {
        if (tid < s) sd[tid] += sd[tid + s];
        __syncthreads();
    }
    if (tid == 0) maxval[q] = (logf(sd[0]) + m10) * 0.1f;
}

// ---------------------------------------------------------------- final gate
__global__ void __launch_bounds__(256)
final_kernel(const float* __restrict__ x, const float* __restrict__ refined,
             const float* __restrict__ thresh, const float* __restrict__ temp_raw,
             const float* __restrict__ alpha, const float* __restrict__ maxval,
             float* __restrict__ out) {
    int idx = blockIdx.x * 256 + threadIdx.x;
    int plane = idx >> 16, c = plane & 15;
    float r = refined[idx];
    float dyn_t = sigm(thresh[c]) * maxval[plane];
    float tau = 1.0f + 19.0f * sigm(temp_raw[0]);
    float gate = sigm(tau * (r - dyn_t));
    out[idx] = x[idx] + sigm(alpha[0]) * r * gate;
}

// ============================================================================
extern "C" void kernel_launch(void* const* d_in, const int* in_sizes, int n_in,
                              void* d_out, int out_size, void* d_ws, size_t ws_size,
                              hipStream_t stream) {
    const float* x        = (const float*)d_in[0];
    const float* th_scale = (const float*)d_in[1];
    const float* th_bias  = (const float*)d_in[2];
    const float* pos_w    = (const float*)d_in[3];
    const float* neg_w    = (const float*)d_in[4];
    const float* W1       = (const float*)d_in[5];
    const float* mix      = (const float*)d_in[6];
    const float* beta_raw = (const float*)d_in[7];
    const float* W2       = (const float*)d_in[8];
    const float* thresh   = (const float*)d_in[9];
    const float* temp_raw = (const float*)d_in[10];
    const float* alpha    = (const float*)d_in[11];
    float* out = (float*)d_out;

    char* w = (char*)d_ws;
    _Float16* combH = (_Float16*)(w);                       // 4*128*65536 f16 = 64 MiB
    _Float16* edH   = (_Float16*)(w + 67108864);            // 4*32*65536 f16  = 16 MiB
    float*    refnd = (float*)   (w + 83886080);            // 4*16*65536 f32  = 16 MiB
    float*    wsA   = (float*)   (w + 100663296);           // f32 scratch     = 16 MiB
    float*    wsB   = (float*)   (w + 117440512);           // f32 scratch     = 16 MiB
    _Float16* W1h   = (_Float16*)(w + 134217728);           // 16x128 f16
    _Float16* W2h   = (_Float16*)(w + 134221824);           // 16x32 f16
    float*    mv    = (float*)   (w + 134222848);           // 64 f32

    const dim3 EB(256), EG(NTOT / 256);                     // elementwise grids
    const dim3 GB(128), GG(4096);                           // GEMM: 64 px/block

    // host-computed Gaussian taps (sigmas 1.0, 1.2, 1.2)
    const float sigmas[3] = {1.0f, 1.2f, 1.2f};
    G5 G[3];
    for (int i = 0; i < 3; ++i) {
        float c[5], s = 0.0f;
        for (int j = 0; j < 5; ++j) {
            float d = (float)j - 2.0f;
            c[j] = expf(-d * d / (2.0f * sigmas[i] * sigmas[i]));
            s += c[j];
        }
        for (int j = 0; j < 5; ++j) G[i].g[j] = c[j] / s;
    }

    prep_weights_kernel<<<10, 256, 0, stream>>>(W1, W2, W1h, W2h);
    prep_x_kernel<<<EG, EB, 0, stream>>>(x, combH);

    const int ks[3] = {3, 5, 7};
    for (int i = 0; i < 3; ++i) {
        erode_kernel<<<EG, EB, 0, stream>>>(x, wsA, ks[i]);
        tophat_kernel<<<EG, EB, 0, stream>>>(x, wsA, th_scale, th_bias, combH, i, ks[i]);
    }
    for (int i = 0; i < 3; ++i) {
        gaussh_kernel<<<EG, EB, 0, stream>>>(x, wsA, G[i]);
        gaussv_kernel<<<EG, EB, 0, stream>>>(wsA, wsB, G[i]);
        log_kernel<<<EG, EB, 0, stream>>>(wsB, pos_w, neg_w, combH, i);
    }

    w1_gemm_wmma<<<GG, GB, 0, stream>>>(combH, W1h, edH);
    median_kernel<<<EG, EB, 0, stream>>>(edH, mix, beta_raw, edH);
    w2_gemm_wmma<<<GG, GB, 0, stream>>>(edH, W2h, refnd);
    lse_kernel<<<64, 256, 0, stream>>>(refnd, mv);
    final_kernel<<<EG, EB, 0, stream>>>(x, refnd, thresh, temp_raw, alpha, mv, out);
}
